// Model_43800076485500
// MI455X (gfx1250) — matrix operand order repaired and checked by reading, not compiled, not run on hardware
//
#include <hip/hip_runtime.h>
#include <hip/hip_bf16.h>
#include <math.h>

// ------------------------------------------------------------------
// CDNA5 WMMA types (wave32): v_wmma_f32_16x16x32_f16
// ------------------------------------------------------------------
typedef _Float16 v16h __attribute__((ext_vector_type(16)));
typedef float    v8f  __attribute__((ext_vector_type(8)));
typedef _Float16 h2v  __attribute__((ext_vector_type(2)));
typedef _Float16 h4v  __attribute__((ext_vector_type(4)));

#define D_MODEL 512
#define N_HEADS 8
#define DK      64
#define D_FF    2048
#define N_LAYERS 4
#define BBATCH  4
#define S_IN    512
#define S_OUT   512
#define VOCAB   32000
#define NEGV    (-1e18f)

#define BK   32
#define LDSK 40   // padded half-stride: 80B -> b64-aligned rows, conflict-free frag reads

// ------------------------------------------------------------------
// Async global->LDS copy (ISA 15.18.3: GLOBAL_LOAD_ASYNC_TO_LDS_B64,
// tracked by ASYNCcnt; S_WAIT_ASYNCCNT to retire). LDS operand is the
// wave-relative byte address = low 32 bits of the generic pointer.
// ------------------------------------------------------------------
__device__ __forceinline__ void async_b64(unsigned int lds_off, const void* gaddr) {
  asm volatile("global_load_async_to_lds_b64 %0, %1, off"
               :: "v"(lds_off), "v"(gaddr) : "memory");
}
__device__ __forceinline__ void wait_async8() {
  asm volatile("s_wait_asynccnt 8" ::: "memory");
}
__device__ __forceinline__ void wait_async0() {
  asm volatile("s_wait_asynccnt 0" ::: "memory");
}

// ------------------------------------------------------------------
// Fragment loaders per ISA 7.12.2 f16 VGPR layouts (wave32).
// A 16x32: lanes 0-15 = rows; VGPR i<4: K=half*8+2i(+w); i>=4: K=16+half*8+2(i-4)(+w)
// B 32x16: lanes 0-15 = cols; VGPR i<4: K=half*8+2i(+w); i>=4: K=16+half*8+2(i-4)(+w)
// Aligned f16-pair reads -> ds_load_2addr_b32.
// ------------------------------------------------------------------
__device__ __forceinline__ v16h load_a_frag(const _Float16* As, int row0, int lane) {
  const int hl = (lane >> 4) & 1, m = lane & 15;
  const _Float16* rp = As + (row0 + m) * LDSK;
  v16h f;
#pragma unroll
  for (int vg = 0; vg < 8; ++vg) {
    int kp = (vg < 4) ? (hl * 8 + 2 * vg) : (16 + hl * 8 + 2 * (vg - 4));
    h2v p = *(const h2v*)(rp + kp);
    f[2 * vg] = p[0]; f[2 * vg + 1] = p[1];
  }
  return f;
}
__device__ __forceinline__ v16h load_b_frag(const _Float16* Bs, int col0, int lane) {
  const int hl = (lane >> 4) & 1, n = lane & 15;
  const _Float16* rp = Bs + (col0 + n) * LDSK;
  v16h f;
#pragma unroll
  for (int vg = 0; vg < 8; ++vg) {
    int kp = (vg < 4) ? (hl * 8 + 2 * vg) : (16 + hl * 8 + 2 * (vg - 4));
    h2v p = *(const h2v*)(rp + kp);
    f[2 * vg] = p[0]; f[2 * vg + 1] = p[1];
  }
  return f;
}

// ------------------------------------------------------------------
// Strided-batched WMMA GEMM, both operands pre-converted f16:
//   A: [M,K] row-major (lda), B: [N,K] row-major (ldb)  ->  C(f32) [M,N] (ldc)
//   C = act(alpha * A x B^T + bias)
// Double-buffered LDS tiles filled with global_load_async_to_lds_b64,
// pipelined one k-step ahead (ASYNCcnt wait <= per-stage issue count).
// Batch z = (b = z/Hn, h = z%Hn) with independent element offsets.
// ------------------------------------------------------------------
template <int BMt, int BNt, int WR, int WC>
__global__ __launch_bounds__(WR* WC * 32) void gemm_wmma(
    const _Float16* __restrict__ A, const _Float16* __restrict__ B,
    const float* __restrict__ bias, float* __restrict__ C,
    int M, int N, int K, int lda, int ldb, int ldc,
    long oAb, long oAh, long oBb, long oBh, long oCb, long oCh,
    int Hn, float alpha, int relu)
{
  constexpr int T  = WR * WC * 32;
  constexpr int MT = BMt / WR / 16;
  constexpr int NT = BNt / WC / 16;
  __shared__ _Float16 As[2][BMt * LDSK];
  __shared__ _Float16 Bs[2][BNt * LDSK];

  const int z = blockIdx.z, bb = z / Hn, hh = z % Hn;
  A += (size_t)bb * oAb + (size_t)hh * oAh;
  B += (size_t)bb * oBb + (size_t)hh * oBh;
  C += (size_t)bb * oCb + (size_t)hh * oCh;

  const int tid = threadIdx.x, lane = tid & 31, wave = tid >> 5;
  const int wr = wave / WC, wc = wave % WC;
  const int m0 = blockIdx.y * BMt, n0 = blockIdx.x * BNt;

  unsigned int aB[2], bB[2];
  aB[0] = (unsigned int)(size_t)&As[0][0]; aB[1] = (unsigned int)(size_t)&As[1][0];
  bB[0] = (unsigned int)(size_t)&Bs[0][0]; bB[1] = (unsigned int)(size_t)&Bs[1][0];

  auto stage = [&](int k0, int buf) {
#pragma unroll
    for (int i = 0; i < BMt * 8 / T; ++i) {           // A tile: BMt rows x 32 halves
      int id = tid + i * T, r = id >> 3, c = id & 7;  // 8x b64 chunks per row
      async_b64(aB[buf] + (unsigned int)(r * LDSK + c * 4) * 2,
                A + (size_t)(m0 + r) * lda + k0 + c * 4);
    }
#pragma unroll
    for (int i = 0; i < BNt * 8 / T; ++i) {           // B tile: BNt rows x 32 halves
      int id = tid + i * T, r = id >> 3, c = id & 7;
      async_b64(bB[buf] + (unsigned int)(r * LDSK + c * 4) * 2,
                B + (size_t)(n0 + r) * ldb + k0 + c * 4);
    }
  };

  v8f acc[MT][NT];
#pragma unroll
  for (int i = 0; i < MT; ++i)
#pragma unroll
    for (int j = 0; j < NT; ++j) acc[i][j] = v8f{0.f,0.f,0.f,0.f,0.f,0.f,0.f,0.f};

  const int nK = K / BK;
  stage(0, 0);
  for (int t = 0; t < nK; ++t) {
    if (t + 1 < nK) { stage((t + 1) * BK, (t + 1) & 1); wait_async8(); }
    else           { wait_async0(); }
    __syncthreads();

    const _Float16* Ab = &As[t & 1][0];
    const _Float16* Bb = &Bs[t & 1][0];
    v16h af[MT], bf[NT];
#pragma unroll
    for (int i = 0; i < MT; ++i) af[i] = load_a_frag(Ab, wr * MT * 16 + i * 16, lane);
#pragma unroll
    for (int j = 0; j < NT; ++j) bf[j] = load_b_frag(Bb, wc * NT * 16 + j * 16, lane);
#pragma unroll
    for (int i = 0; i < MT; ++i)
#pragma unroll
      for (int j = 0; j < NT; ++j)
        acc[i][j] = __builtin_amdgcn_wmma_f32_16x16x32_f16(
            false, af[i], false, bf[j], (short)0, acc[i][j], false, false);
    __syncthreads();
  }

  // C/D layout: lane = col (0-15 per half), VGPR r -> row r + 8*half.
  const int hl = lane >> 4, nl = lane & 15;
  const int mW = m0 + wr * MT * 16, nW = n0 + wc * NT * 16;
#pragma unroll
  for (int i = 0; i < MT; ++i)
#pragma unroll
    for (int j = 0; j < NT; ++j) {
      int col = nW + j * 16 + nl;
      float bv = bias ? bias[col] : 0.f;
#pragma unroll
      for (int r = 0; r < 8; ++r) {
        int row = mW + i * 16 + hl * 8 + r;
        float v = alpha * acc[i][j][r] + bv;
        if (relu) v = fmaxf(v, 0.f);
        C[(size_t)row * ldc + col] = v;
      }
    }
}

// ------------------------------------------------------------------
// f32 -> f16 flat copy (vectorized: float4 in, 4xf16 out)
// ------------------------------------------------------------------
__global__ void hcopy_kernel(const float* __restrict__ in, _Float16* __restrict__ out, long n4)
{
  long i = (long)blockIdx.x * blockDim.x + threadIdx.x;
  if (i < n4) {
    float4 v = ((const float4*)in)[i];
    h4v h = { (_Float16)v.x, (_Float16)v.y, (_Float16)v.z, (_Float16)v.w };
    ((h4v*)out)[i] = h;
  }
}

// ------------------------------------------------------------------
// Batched f32 [R,C] -> f16 [C,R] transpose via padded LDS tile.
// ------------------------------------------------------------------
__global__ void transpose_h_kernel(const float* __restrict__ in, _Float16* __restrict__ out,
                                   int ldi, int ldo,
                                   long oIb, long oIh, long oOb, long oOh, int Hn)
{
  __shared__ float tile[32][33];
  const int z = blockIdx.z, bb = z / Hn, hh = z % Hn;
  in  += (size_t)bb * oIb + (size_t)hh * oIh;
  out += (size_t)bb * oOb + (size_t)hh * oOh;
  const int r0 = blockIdx.y * 32, c0 = blockIdx.x * 32;
  const int tx = threadIdx.x & 31, ty = threadIdx.x >> 5;  // 32 x 8
#pragma unroll
  for (int i = 0; i < 4; ++i)
    tile[ty + 8 * i][tx] = in[(size_t)(r0 + ty + 8 * i) * ldi + c0 + tx];
  __syncthreads();
#pragma unroll
  for (int i = 0; i < 4; ++i)
    out[(size_t)(c0 + ty + 8 * i) * ldo + r0 + tx] = (_Float16)tile[tx][ty + 8 * i];
}

// ------------------------------------------------------------------
// Row softmax, mask built from token indices (pad + optional causal).
// ------------------------------------------------------------------
__global__ void mask_softmax_kernel(float* __restrict__ scores, const int* __restrict__ kidx,
                                    int Sq, int Sk, int Hn, int causal)
{
  const int q = blockIdx.x, z = blockIdx.y, b = z / Hn;
  float* row = scores + ((size_t)z * Sq + q) * Sk;
  const int* kp = kidx + (size_t)b * Sk;
  const int tid = threadIdx.x;
  __shared__ float red[256];

  float vals[4];
  const int nIt = (Sk + 255) >> 8;
  float mx = -3.0e38f;
  for (int i = 0; i < nIt; ++i) {
    int j = tid + (i << 8);
    float v = -3.0e38f;
    if (j < Sk) {
      bool keep = (kp[j] != 0) && (!causal || (j <= q));
      v = keep ? row[j] : NEGV;
    }
    vals[i] = v; mx = fmaxf(mx, v);
  }
  red[tid] = mx; __syncthreads();
  for (int s = 128; s > 0; s >>= 1) { if (tid < s) red[tid] = fmaxf(red[tid], red[tid + s]); __syncthreads(); }
  mx = red[0]; __syncthreads();

  float sum = 0.f;
  for (int i = 0; i < nIt; ++i) {
    int j = tid + (i << 8);
    if (j < Sk) { vals[i] = __expf(vals[i] - mx); sum += vals[i]; }
  }
  red[tid] = sum; __syncthreads();
  for (int s = 128; s > 0; s >>= 1) { if (tid < s) red[tid] += red[tid + s]; __syncthreads(); }
  const float inv = 1.f / red[0];
  for (int i = 0; i < nIt; ++i) {
    int j = tid + (i << 8);
    if (j < Sk) row[j] = vals[i] * inv;
  }
}

// out = LayerNorm(a + res) * g + beta (row-wise over D)
__global__ void add_ln_kernel(const float* __restrict__ a, const float* __restrict__ res,
                              const float* __restrict__ g, const float* __restrict__ beta,
                              float* __restrict__ out, int D)
{
  const int row = blockIdx.x, tid = threadIdx.x;
  __shared__ float red[256];
  float vals[4];
  const int nIt = (D + 255) >> 8;
  float s = 0.f;
  for (int i = 0; i < nIt; ++i) {
    int j = tid + (i << 8);
    float v = 0.f;
    if (j < D) v = a[(size_t)row * D + j] + res[(size_t)row * D + j];
    vals[i] = v; s += v;
  }
  red[tid] = s; __syncthreads();
  for (int t = 128; t > 0; t >>= 1) { if (tid < t) red[tid] += red[tid + t]; __syncthreads(); }
  const float mu = red[0] / (float)D; __syncthreads();
  float vs = 0.f;
  for (int i = 0; i < nIt; ++i) {
    int j = tid + (i << 8);
    if (j < D) { float d = vals[i] - mu; vs += d * d; }
  }
  red[tid] = vs; __syncthreads();
  for (int t = 128; t > 0; t >>= 1) { if (tid < t) red[tid] += red[tid + t]; __syncthreads(); }
  const float rstd = rsqrtf(red[0] / (float)D + 1e-5f);
  for (int i = 0; i < nIt; ++i) {
    int j = tid + (i << 8);
    if (j < D) out[(size_t)row * D + j] = (vals[i] - mu) * rstd * g[j] + beta[j];
  }
}

// x[t,:] = word_emb[idx[t],:] + pos_emb[idx[t],:]
__global__ void embed_kernel(const int* __restrict__ idx, const float* __restrict__ we,
                             const float* __restrict__ pe, float* __restrict__ x, int D)
{
  const int t = blockIdx.x;
  const int tok = idx[t];
  for (int d = threadIdx.x; d < D; d += blockDim.x)
    x[(size_t)t * D + d] = we[(size_t)tok * D + d] + pe[(size_t)tok * D + d];
}

// ------------------------------------------------------------------
// Host orchestration
// ------------------------------------------------------------------
struct AttnP { const float *wq,*bq,*wk,*bk,*wv,*bv,*wo,*bo,*g,*b; };

static void hcopy(hipStream_t st, const float* in, _Float16* out, long n) {
  long n4 = n / 4;
  hcopy_kernel<<<dim3((unsigned)((n4 + 255) / 256)), dim3(256), 0, st>>>(in, out, n4);
}
static void transpose_h(hipStream_t st, const float* in, _Float16* out, int R, int C,
                        int ldi, int ldo, int batch = 1, int Hn = 1,
                        long oIb = 0, long oIh = 0, long oOb = 0, long oOh = 0) {
  transpose_h_kernel<<<dim3(C / 32, R / 32, batch), dim3(256), 0, st>>>(
      in, out, ldi, ldo, oIb, oIh, oOb, oOh, Hn);
}
static void gemm128(hipStream_t st, const _Float16* A, const _Float16* B, const float* bias,
                    float* C, int M, int N, int K, int lda, int ldb, int ldc,
                    float alpha = 1.f, int relu = 0, int batch = 1, int Hn = 1,
                    long oAb = 0, long oAh = 0, long oBb = 0, long oBh = 0, long oCb = 0, long oCh = 0) {
  gemm_wmma<128, 128, 4, 2><<<dim3(N / 128, M / 128, batch), dim3(256), 0, st>>>(
      A, B, bias, C, M, N, K, lda, ldb, ldc, oAb, oAh, oBb, oBh, oCb, oCh, Hn, alpha, relu);
}
static void gemm64(hipStream_t st, const _Float16* A, const _Float16* B, const float* bias,
                   float* C, int M, int N, int K, int lda, int ldb, int ldc,
                   float alpha = 1.f, int relu = 0, int batch = 1, int Hn = 1,
                   long oAb = 0, long oAh = 0, long oBb = 0, long oBh = 0, long oCb = 0, long oCh = 0) {
  gemm_wmma<64, 64, 2, 2><<<dim3(N / 64, M / 64, batch), dim3(128), 0, st>>>(
      A, B, bias, C, M, N, K, lda, ldb, ldc, oAb, oAh, oBb, oBh, oCb, oCh, Hn, alpha, relu);
}

struct Ws {
  float *q, *kbuf, *vbuf, *ctx, *t0, *xenc, *ydec, *hff, *sc;
  _Float16 *Ah, *Bh;
};

static void mha(hipStream_t st, Ws& w, float* io_q, const float* kv, int Sq, int Sk,
                const AttnP& p, const int* kidx, int causal)
{
  const int Mq = BBATCH * Sq, Mk = BBATCH * Sk;
  const int BH = BBATCH * N_HEADS;

  // ---- Q/K/V projections: A = activations(f16), B = W^T(f16 [N,K]) ----
  hcopy(st, io_q, w.Ah, (long)Mq * D_MODEL);
  transpose_h(st, p.wq, w.Bh, D_MODEL, D_MODEL, D_MODEL, D_MODEL);
  gemm128(st, w.Ah, w.Bh, p.bq, w.q, Mq, D_MODEL, D_MODEL, D_MODEL, D_MODEL, D_MODEL);
  if (kv != io_q) hcopy(st, kv, w.Ah, (long)Mk * D_MODEL);
  transpose_h(st, p.wk, w.Bh, D_MODEL, D_MODEL, D_MODEL, D_MODEL);
  gemm128(st, w.Ah, w.Bh, p.bk, w.kbuf, Mk, D_MODEL, D_MODEL, D_MODEL, D_MODEL, D_MODEL);
  transpose_h(st, p.wv, w.Bh, D_MODEL, D_MODEL, D_MODEL, D_MODEL);
  gemm128(st, w.Ah, w.Bh, p.bv, w.vbuf, Mk, D_MODEL, D_MODEL, D_MODEL, D_MODEL, D_MODEL);

  // ---- scores[b,h] = (1/sqrt(dk)) Q K^T : K rows are key-major => already [N,K] ----
  hcopy(st, w.q, w.Ah, (long)Mq * D_MODEL);
  hcopy(st, w.kbuf, w.Bh, (long)Mk * D_MODEL);
  gemm128(st, w.Ah, w.Bh, nullptr, w.sc, Sq, Sk, DK, D_MODEL, D_MODEL, Sk,
          0.125f, 0, BH, N_HEADS,
          (long)Sq * D_MODEL, DK,
          (long)Sk * D_MODEL, DK,
          (long)N_HEADS * Sq * Sk, (long)Sq * Sk);
  mask_softmax_kernel<<<dim3(Sq, BH), dim3(256), 0, st>>>(w.sc, kidx, Sq, Sk, N_HEADS, causal);

  // ---- ctx[b,h] = attn x V : V transposed per-(b,h) to [d, key] f16 ----
  hcopy(st, w.sc, w.Ah, (long)BH * Sq * Sk);
  transpose_h(st, w.vbuf, w.Bh, Sk, DK, D_MODEL, Sk, BH, N_HEADS,
              (long)Sk * D_MODEL, DK,
              (long)N_HEADS * DK * Sk, (long)DK * Sk);
  gemm64(st, w.Ah, w.Bh, nullptr, w.ctx, Sq, DK, Sk, Sk, Sk, D_MODEL,
         1.f, 0, BH, N_HEADS,
         (long)N_HEADS * Sq * Sk, (long)Sq * Sk,
         (long)N_HEADS * DK * Sk, (long)DK * Sk,
         (long)Sq * D_MODEL, DK);

  // ---- output projection + residual LayerNorm ----
  hcopy(st, w.ctx, w.Ah, (long)Mq * D_MODEL);
  transpose_h(st, p.wo, w.Bh, D_MODEL, D_MODEL, D_MODEL, D_MODEL);
  gemm128(st, w.Ah, w.Bh, p.bo, w.t0, Mq, D_MODEL, D_MODEL, D_MODEL, D_MODEL, D_MODEL);
  add_ln_kernel<<<dim3(Mq), dim3(256), 0, st>>>(w.t0, io_q, p.g, p.b, io_q, D_MODEL);
}

static void ffn(hipStream_t st, Ws& w, float* x,
                const float* f1w, const float* f1b, const float* f2w, const float* f2b,
                const float* g, const float* b, int M)
{
  hcopy(st, x, w.Ah, (long)M * D_MODEL);
  transpose_h(st, f1w, w.Bh, D_MODEL, D_FF, D_FF, D_MODEL);
  gemm128(st, w.Ah, w.Bh, f1b, w.hff, M, D_FF, D_MODEL, D_MODEL, D_MODEL, D_FF, 1.f, 1);
  hcopy(st, w.hff, w.Ah, (long)M * D_FF);
  transpose_h(st, f2w, w.Bh, D_FF, D_MODEL, D_MODEL, D_FF);
  gemm128(st, w.Ah, w.Bh, f2b, w.t0, M, D_MODEL, D_FF, D_FF, D_FF, D_MODEL);
  add_ln_kernel<<<dim3(M), dim3(256), 0, st>>>(w.t0, x, g, b, x, D_MODEL);
}

extern "C" void kernel_launch(void* const* d_in, const int* in_sizes, int n_in,
                              void* d_out, int out_size, void* d_ws, size_t ws_size,
                              hipStream_t stream)
{
  (void)in_sizes; (void)n_in; (void)out_size; (void)ws_size;

  // Depth-first insertion-order flattening of setup_inputs():
  // 0 in_idx, 1 out_idx, 2..5 embeddings,
  // enc sa 6..15 (wq,bq,wk,bk,wv,bv,wo,bo,ln_g,ln_b), enc ffn 16..21,
  // dec sa 22..31, dec ma 32..41, dec ffn 42..47, 48 out_w, 49 out_b.
  const int*   in_idx  = (const int*)d_in[0];
  const int*   out_idx = (const int*)d_in[1];
  const float* ewe = (const float*)d_in[2];
  const float* epe = (const float*)d_in[3];
  const float* dwe = (const float*)d_in[4];
  const float* dpe = (const float*)d_in[5];
  const float* out_w = (const float*)d_in[48];
  const float* out_b = (const float*)d_in[49];

  Ws w;
  float* p = (float*)d_ws;
  const size_t MS = (size_t)BBATCH * S_IN * D_MODEL;       // 1,048,576
  w.q    = p; p += MS;
  w.kbuf = p; p += MS;
  w.vbuf = p; p += MS;
  w.ctx  = p; p += MS;
  w.t0   = p; p += MS;
  w.xenc = p; p += MS;
  w.ydec = p; p += MS;
  w.hff  = p; p += (size_t)BBATCH * S_IN * D_FF;           // 4,194,304
  w.sc   = p; p += (size_t)BBATCH * N_HEADS * S_IN * S_IN; // 8,388,608
  w.Ah   = (_Float16*)p;                                    // 8,388,608 halves
  w.Bh   = w.Ah + (size_t)BBATCH * N_HEADS * S_IN * S_IN;   // 16,384,000 halves

  embed_kernel<<<dim3(BBATCH * S_IN),  dim3(256), 0, stream>>>(in_idx,  ewe, epe, w.xenc, D_MODEL);
  embed_kernel<<<dim3(BBATCH * S_OUT), dim3(256), 0, stream>>>(out_idx, dwe, dpe, w.ydec, D_MODEL);

  auto attP = [&](int base, int layer) {
    AttnP a;
    a.wq = (const float*)d_in[base + 0] + (size_t)layer * D_MODEL * D_MODEL;
    a.bq = (const float*)d_in[base + 1] + (size_t)layer * D_MODEL;
    a.wk = (const float*)d_in[base + 2] + (size_t)layer * D_MODEL * D_MODEL;
    a.bk = (const float*)d_in[base + 3] + (size_t)layer * D_MODEL;
    a.wv = (const float*)d_in[base + 4] + (size_t)layer * D_MODEL * D_MODEL;
    a.bv = (const float*)d_in[base + 5] + (size_t)layer * D_MODEL;
    a.wo = (const float*)d_in[base + 6] + (size_t)layer * D_MODEL * D_MODEL;
    a.bo = (const float*)d_in[base + 7] + (size_t)layer * D_MODEL;
    a.g  = (const float*)d_in[base + 8] + (size_t)layer * D_MODEL;
    a.b  = (const float*)d_in[base + 9] + (size_t)layer * D_MODEL;
    return a;
  };
  auto ffnRun = [&](int base, int layer, float* x, int M) {
    const float* f1w = (const float*)d_in[base + 0] + (size_t)layer * D_MODEL * D_FF;
    const float* f1b = (const float*)d_in[base + 1] + (size_t)layer * D_FF;
    const float* f2w = (const float*)d_in[base + 2] + (size_t)layer * D_FF * D_MODEL;
    const float* f2b = (const float*)d_in[base + 3] + (size_t)layer * D_MODEL;
    const float* lg  = (const float*)d_in[base + 4] + (size_t)layer * D_MODEL;
    const float* lb  = (const float*)d_in[base + 5] + (size_t)layer * D_MODEL;
    ffn(stream, w, x, f1w, f1b, f2w, f2b, lg, lb, M);
  };

  // ---------------- encoder ----------------
  for (int l = 0; l < N_LAYERS; ++l) {
    AttnP sa = attP(6, l);
    mha(stream, w, w.xenc, w.xenc, S_IN, S_IN, sa, in_idx, /*causal=*/0);
    ffnRun(16, l, w.xenc, BBATCH * S_IN);
  }
  // ---------------- decoder ----------------
  for (int l = 0; l < N_LAYERS; ++l) {
    AttnP sa = attP(22, l);
    mha(stream, w, w.ydec, w.ydec, S_OUT, S_OUT, sa, out_idx, /*causal=*/1);
    AttnP ma = attP(32, l);
    mha(stream, w, w.ydec, w.xenc, S_OUT, S_IN, ma, in_idx, /*causal=*/0);
    ffnRun(42, l, w.ydec, BBATCH * S_OUT);
  }
  // ---------------- logits: [2048,512] x [512,32000] + bias ----------------
  hcopy(stream, w.ydec, w.Ah, (long)BBATCH * S_OUT * D_MODEL);
  transpose_h(stream, out_w, w.Bh, D_MODEL, VOCAB, VOCAB, D_MODEL);
  gemm128(stream, w.Ah, w.Bh, out_b, (float*)d_out,
          BBATCH * S_OUT, VOCAB, D_MODEL, D_MODEL, D_MODEL, VOCAB);
}
